// STAR_24670292148507
// MI455X (gfx1250) — matrix operand order repaired and checked by reading, not compiled, not run on hardware
//
#include <hip/hip_runtime.h>
#include <math.h>

// ---------------- model constants ----------------
constexpr int BB   = 16;            // batch
constexpr int CH   = 3;
constexpr int IMG  = 224;
constexpr int PP   = 16;            // patch
constexpr int DD   = 192;           // model dim
constexpr int DEPTH= 12;
constexpr int EE   = 384;           // mamba inner
constexpr int NSs  = 16;            // state size
constexpr int RR   = 12;            // dt rank
constexpr int KC   = 4;             // conv kernel
constexpr int HEADS= 3;
constexpr int NCLS = 1000;
constexpr int GG   = IMG / PP;      // 14
constexpr int NPp  = GG * GG;       // 196
constexpr int LL   = NPp + 1;       // 197
constexpr int MIDt = NPp / 2;       // 98
constexpr int HDd  = DD / HEADS;    // 64
constexpr int FF   = 4 * DD;        // 768
constexpr int BL   = BB * LL;       // 3152
constexpr int DBLW = RR + 2 * NSs;  // 44

// ---------------- workspace layout (floats) ----------------
constexpr size_t SZ_TOK = (size_t)BL * DD;       // 605184
constexpr size_t SZ_XZ  = (size_t)BL * 2 * EE;   // 2420736
constexpr size_t SZ_XS  = (size_t)BL * EE;       // 1210368
constexpr size_t SZ_DBL = (size_t)BL * DBLW;     // 138688
constexpr size_t O_HID = 0;
constexpr size_t O_RES = O_HID + SZ_TOK;
constexpr size_t O_HN  = O_RES + SZ_TOK;
constexpr size_t O_XZ  = O_HN  + SZ_TOK;
constexpr size_t O_XS  = O_XZ  + SZ_XZ;
constexpr size_t O_DBL = O_XS  + SZ_XS;
constexpr size_t O_DT  = O_DBL + SZ_DBL;
constexpr size_t O_YF  = O_DT  + SZ_XS;
constexpr size_t O_YG  = O_YF  + SZ_XS;
// total = O_YG + SZ_XS = 9,216,448 floats (~36.9 MB) -- must fit in ws

typedef __attribute__((ext_vector_type(16))) _Float16 v16h;
typedef __attribute__((ext_vector_type(8)))  float    v8f;

// ---------------- device helpers ----------------
__device__ __forceinline__ float silu_f(float x) {
  return x * (1.0f / (1.0f + __expf(-x)));
}
__device__ __forceinline__ float gelu_tanh(float x) {
  const float c = 0.7978845608028654f; // sqrt(2/pi)
  return 0.5f * x * (1.0f + tanhf(c * (x + 0.044715f * x * x * x)));
}
__device__ __forceinline__ float softplus_f(float x) {
  return (x > 20.0f) ? x : log1pf(__expf(x));
}
__device__ __forceinline__ float wave_sum(float v) {
  #pragma unroll
  for (int off = 16; off; off >>= 1) v += __shfl_xor(v, off, 32);
  return v;
}

// ---------------- WMMA GEMM:  Y[M,N] = epi(A[M,K](lda) * W[N,K]^T) ----------
// One wave owns a 16x64 output strip: one shared A-fragment, 4 B-fragments,
// 4 f32 accumulators -> 4 v_wmma_f32_16x16x32_f16 per K-chunk (A reuse in
// registers; 5 b128 loads per WMMA).  epi: +bias, act (0 none / 1 gelu /
// 2 softplus), +resid.  M multiple of 16; N, K arbitrary (guard/zero-pad:
// EXEC stays all-ones at every WMMA).
__global__ void k_gemm(const float* __restrict__ A, int lda,
                       const float* __restrict__ W,
                       const float* __restrict__ bias,
                       const float* __restrict__ resid,
                       float* __restrict__ Y,
                       int M, int N, int K, int act)
{
  const int wid  = threadIdx.x >> 5;
  const int lane = threadIdx.x & 31;
  const int ng   = (N + 63) >> 6;                 // 64-wide column groups
  const int tile = blockIdx.x * (blockDim.x >> 5) + wid;
  if (tile >= (M >> 4) * ng) return;
  const int tm = tile / ng, tg = tile % ng;
  const int row0 = tm << 4, colb = tg << 6;
  const int m = lane & 15, h = lane >> 4;
  const float* __restrict__ Arow = A + (size_t)(row0 + m) * lda;
  const float* __restrict__ Wrow[4];
  int ncol[4];
  #pragma unroll
  for (int j = 0; j < 4; ++j) {
    const int n = colb + (j << 4) + m;
    ncol[j] = n;
    Wrow[j] = W + (size_t)((n < N) ? n : (N - 1)) * K;
  }

  v8f c0 = {}, c1 = {}, c2 = {}, c3 = {};
  for (int k0 = 0; k0 < K; k0 += 32) {
    v16h a, b0, b1, b2, b3;
    if (k0 + 32 <= K) {
      if (k0 + 64 <= K) {
        __builtin_prefetch((const void*)(Arow + k0 + 32), 0, 3);
        __builtin_prefetch((const void*)(Wrow[0] + k0 + 32), 0, 3);
        __builtin_prefetch((const void*)(Wrow[1] + k0 + 32), 0, 3);
        __builtin_prefetch((const void*)(Wrow[2] + k0 + 32), 0, 3);
        __builtin_prefetch((const void*)(Wrow[3] + k0 + 32), 0, 3);
      }
      #pragma unroll
      for (int v = 0; v < 8; ++v) {
        // A 16x32 f16 layout: lane m, half h; vgpr v holds one K pair
        const int ka = k0 + (v << 1) + (h << 4);
        a[2 * v]     = (_Float16)Arow[ka];
        a[2 * v + 1] = (_Float16)Arow[ka + 1];
        // B 32x16 f16, this code's contiguous k order (same as A): lane n; vgpr v holds K pair 2v(+16h)
        const int kb = k0 + (v << 1) + (h << 4);
        b0[2 * v] = (_Float16)Wrow[0][kb]; b0[2 * v + 1] = (_Float16)Wrow[0][kb + 1];
        b1[2 * v] = (_Float16)Wrow[1][kb]; b1[2 * v + 1] = (_Float16)Wrow[1][kb + 1];
        b2[2 * v] = (_Float16)Wrow[2][kb]; b2[2 * v + 1] = (_Float16)Wrow[2][kb + 1];
        b3[2 * v] = (_Float16)Wrow[3][kb]; b3[2 * v + 1] = (_Float16)Wrow[3][kb + 1];
      }
    } else { // K remainder chunk: zero-pad
      #pragma unroll
      for (int v = 0; v < 8; ++v) {
        const int ka = k0 + (v << 1) + (h << 4);
        a[2 * v]     = (ka     < K) ? (_Float16)Arow[ka]     : (_Float16)0.0f;
        a[2 * v + 1] = (ka + 1 < K) ? (_Float16)Arow[ka + 1] : (_Float16)0.0f;
        const int kb = k0 + (v << 1) + (h << 4);
        const bool g0 = kb < K, g1 = kb + 1 < K;
        b0[2 * v] = g0 ? (_Float16)Wrow[0][kb] : (_Float16)0.0f;
        b0[2 * v + 1] = g1 ? (_Float16)Wrow[0][kb + 1] : (_Float16)0.0f;
        b1[2 * v] = g0 ? (_Float16)Wrow[1][kb] : (_Float16)0.0f;
        b1[2 * v + 1] = g1 ? (_Float16)Wrow[1][kb + 1] : (_Float16)0.0f;
        b2[2 * v] = g0 ? (_Float16)Wrow[2][kb] : (_Float16)0.0f;
        b2[2 * v + 1] = g1 ? (_Float16)Wrow[2][kb + 1] : (_Float16)0.0f;
        b3[2 * v] = g0 ? (_Float16)Wrow[3][kb] : (_Float16)0.0f;
        b3[2 * v + 1] = g1 ? (_Float16)Wrow[3][kb + 1] : (_Float16)0.0f;
      }
    }
    c0 = __builtin_amdgcn_wmma_f32_16x16x32_f16(false, a, false, b0, (short)0, c0, false, false);
    c1 = __builtin_amdgcn_wmma_f32_16x16x32_f16(false, a, false, b1, (short)0, c1, false, false);
    c2 = __builtin_amdgcn_wmma_f32_16x16x32_f16(false, a, false, b2, (short)0, c2, false, false);
    c3 = __builtin_amdgcn_wmma_f32_16x16x32_f16(false, a, false, b3, (short)0, c3, false, false);
  }

  v8f cc[4] = {c0, c1, c2, c3};
  #pragma unroll
  for (int j = 0; j < 4; ++j) {
    const int n = ncol[j];
    if (n < N) {
      #pragma unroll
      for (int v = 0; v < 8; ++v) {
        const int row = row0 + v + (h << 3); // f32 C/D: vgpr v -> M = v + 8*h
        float t = cc[j][v];
        if (bias)      t += bias[n];
        if (act == 1)  t = gelu_tanh(t);
        else if (act == 2) t = softplus_f(t);
        if (resid)     t += resid[(size_t)row * N + n];
        Y[(size_t)row * N + n] = t;
      }
    }
  }
}

// ---------------- LayerNorm (D=192), wave per row -------------------------
// Xout (optional) receives X+Add (the updated residual); out = LN(X+Add)*w+b
__global__ void k_ln(const float* __restrict__ X, const float* __restrict__ Add,
                     float* __restrict__ Xout,
                     const float* __restrict__ w, const float* __restrict__ b,
                     float* __restrict__ out, int rows)
{
  const int wid = threadIdx.x >> 5, lane = threadIdx.x & 31;
  const int row = blockIdx.x * (blockDim.x >> 5) + wid;
  if (row >= rows) return;
  const float* x = X + (size_t)row * DD;
  float v[6];
  float s = 0.0f;
  #pragma unroll
  for (int i = 0; i < 6; ++i) {
    const int d = lane + 32 * i;
    float t = x[d];
    if (Add) t += Add[(size_t)row * DD + d];
    v[i] = t; s += t;
  }
  const float mean = wave_sum(s) * (1.0f / DD);
  float vs = 0.0f;
  #pragma unroll
  for (int i = 0; i < 6; ++i) { const float d0 = v[i] - mean; vs += d0 * d0; }
  const float rstd = rsqrtf(wave_sum(vs) * (1.0f / DD) + 1e-5f);
  #pragma unroll
  for (int i = 0; i < 6; ++i) {
    const int d = lane + 32 * i;
    if (Xout) Xout[(size_t)row * DD + d] = v[i];
    out[(size_t)row * DD + d] = (v[i] - mean) * rstd * w[d] + b[d];
  }
}

// ---------------- patch embed: im2col + token assembly --------------------
__global__ void k_im2col(const float* __restrict__ img, float* __restrict__ Ap)
{
  const int idx = blockIdx.x * blockDim.x + threadIdx.x;
  if (idx >= BB * NPp * CH * PP * PP) return;
  const int k = idx % (CH * PP * PP);
  const int r = idx / (CH * PP * PP);
  const int p = r % NPp, b = r / NPp;
  const int gy = p / GG, gx = p % GG;
  const int c = k >> 8, rem = k & 255, ph = rem >> 4, pw = rem & 15;
  Ap[idx] = img[(((size_t)b * CH + c) * IMG + (gy * PP + ph)) * IMG + gx * PP + pw];
}

__global__ void k_assemble(const float* __restrict__ Pout,
                           const float* __restrict__ pos,
                           const float* __restrict__ cls,
                           float* __restrict__ hidden, float* __restrict__ res)
{
  const int idx = blockIdx.x * blockDim.x + threadIdx.x;
  if (idx >= BB * LL * DD) return;
  const int d = idx % DD;
  const int t = (idx / DD) % LL;
  const int b = idx / (DD * LL);
  float val;
  if (t == MIDt) val = cls[d];
  else {
    const int p = (t < MIDt) ? t : (t - 1);
    val = Pout[((size_t)b * NPp + p) * DD + d] + pos[(size_t)p * DD + d];
  }
  hidden[idx] = val;
  res[idx] = 0.0f;
}

// ---------------- causal depthwise conv (K=4) + SiLU ----------------------
__global__ void k_conv_silu(const float* __restrict__ xz,
                            const float* __restrict__ cw,
                            const float* __restrict__ cb,
                            float* __restrict__ xs)
{
  const int idx = blockIdx.x * blockDim.x + threadIdx.x;
  if (idx >= BB * LL * EE) return;
  const int e = idx % EE;
  const int l = (idx / EE) % LL;
  const int b = idx / (EE * LL);
  float acc = cb[e];
  const float* wr = cw + (size_t)e * KC;
  #pragma unroll
  for (int k = 0; k < KC; ++k) {
    const int li = l + k - (KC - 1);
    if (li >= 0) acc += xz[((size_t)(b * LL + li)) * (2 * EE) + e] * wr[k];
  }
  xs[idx] = silu_f(acc);
}

// ---------------- bidirectional selective scan ----------------------------
// one thread per (b,e) channel; 16-state SSM kept in registers; forward pass
// stores yf, backward pass fuses yf+yb+x*D and the silu(z) gate -> yg.
__global__ void k_scan(const float* __restrict__ dtb, const float* __restrict__ xs,
                       const float* __restrict__ dbl, const float* __restrict__ xz,
                       const float* __restrict__ A_log, const float* __restrict__ Dp,
                       float* __restrict__ yf, float* __restrict__ yg)
{
  const int b = blockIdx.x;
  const int e = threadIdx.x;        // blockDim == EE
  float A[NSs], h[NSs];
  #pragma unroll
  for (int n = 0; n < NSs; ++n) { A[n] = -__expf(A_log[(size_t)e * NSs + n]); h[n] = 0.0f; }
  for (int l = 0; l < LL; ++l) {    // forward
    const size_t ble = ((size_t)b * LL + l) * EE + e;
    const float dtv = dtb[ble], xv = xs[ble], s = dtv * xv;
    const float* dd = dbl + ((size_t)b * LL + l) * DBLW;
    float y = 0.0f;
    #pragma unroll
    for (int n = 0; n < NSs; ++n) {
      h[n] = __expf(dtv * A[n]) * h[n] + s * dd[RR + n];
      y += h[n] * dd[RR + NSs + n];
    }
    yf[ble] = y;
  }
  #pragma unroll
  for (int n = 0; n < NSs; ++n) h[n] = 0.0f;
  const float dpe = Dp[e];
  for (int l = LL - 1; l >= 0; --l) { // backward + combine + gate
    const size_t ble = ((size_t)b * LL + l) * EE + e;
    const float dtv = dtb[ble], xv = xs[ble], s = dtv * xv;
    const float* dd = dbl + ((size_t)b * LL + l) * DBLW;
    float y = 0.0f;
    #pragma unroll
    for (int n = 0; n < NSs; ++n) {
      h[n] = __expf(dtv * A[n]) * h[n] + s * dd[RR + n];
      y += h[n] * dd[RR + NSs + n];
    }
    const float zv = xz[((size_t)b * LL + l) * (2 * EE) + EE + e];
    yg[ble] = (yf[ble] + y + xv * dpe) * silu_f(zv);
  }
}

// ---------------- attention: one wave per (b, head, query row) ------------
// online (flash-style) softmax; HD=64 -> 2 values per lane.
__global__ void k_attn(const float* __restrict__ qb, const float* __restrict__ kvb,
                       const float* __restrict__ mask, float* __restrict__ ob)
{
  const int wid = threadIdx.x >> 5, lane = threadIdx.x & 31;
  const int gw = blockIdx.x * (blockDim.x >> 5) + wid;
  if (gw >= BB * HEADS * LL) return;
  const int qi = gw % LL;
  const int hh = (gw / LL) % HEADS;
  const int b  = gw / (LL * HEADS);
  const float* qp = qb + ((size_t)b * LL + qi) * DD + hh * HDd;
  const float q0 = qp[2 * lane], q1 = qp[2 * lane + 1];
  float m = -3.0e38f, s = 0.0f, a0 = 0.0f, a1 = 0.0f;
  for (int j = 0; j < LL; ++j) {
    const float* kp = kvb + ((size_t)b * LL + j) * (2 * DD) + hh * HDd;
    float d = wave_sum(q0 * kp[2 * lane] + q1 * kp[2 * lane + 1]);
    const float sc = d * 0.125f + mask[(size_t)qi * LL + j]; // HD^-0.5 = 0.125
    const float mn = fmaxf(m, sc);
    const float f  = __expf(m - mn);
    const float p  = __expf(sc - mn);
    const float* vp = kvb + ((size_t)b * LL + j) * (2 * DD) + DD + hh * HDd;
    a0 = a0 * f + p * vp[2 * lane];
    a1 = a1 * f + p * vp[2 * lane + 1];
    s  = s * f + p;
    m  = mn;
  }
  float* op = ob + ((size_t)b * LL + qi) * DD + hh * HDd;
  op[2 * lane] = a0 / s;
  op[2 * lane + 1] = a1 / s;
}

__global__ void k_gather_cls(const float* __restrict__ x, float* __restrict__ xc)
{
  const int idx = blockIdx.x * blockDim.x + threadIdx.x;
  if (idx >= BB * DD) return;
  const int d = idx % DD, b = idx / DD;
  xc[idx] = x[((size_t)b * LL + MIDt) * DD + d];
}

// ---------------- host side ----------------
static inline void gemm(hipStream_t s, const float* A, int lda, const float* W,
                        const float* bias, const float* resid, float* Y,
                        int M, int N, int K, int act)
{
  const int tiles = (M / 16) * ((N + 63) / 64);  // 16x64 strip per wave
  const int blocks = (tiles + 7) / 8;            // 8 waves (256 threads) / block
  k_gemm<<<blocks, 256, 0, s>>>(A, lda, W, bias, resid, Y, M, N, K, act);
}

extern "C" void kernel_launch(void* const* d_in, const int* in_sizes, int n_in,
                              void* d_out, int out_size, void* d_ws, size_t ws_size,
                              hipStream_t stream)
{
  (void)in_sizes; (void)n_in; (void)out_size; (void)ws_size;
  const float* images    = (const float*)d_in[0];
  const float* mask      = (const float*)d_in[1];
  const float* patch_w   = (const float*)d_in[2];
  const float* patch_b   = (const float*)d_in[3];
  const float* pos_embed = (const float*)d_in[4];
  const float* cls_token = (const float*)d_in[5];
  const float* ln_w      = (const float*)d_in[6];
  const float* ln_b      = (const float*)d_in[7];
  const float* in_proj_w = (const float*)d_in[8];
  const float* conv_w    = (const float*)d_in[9];
  const float* conv_b    = (const float*)d_in[10];
  const float* x_proj_w  = (const float*)d_in[11];
  const float* dt_proj_w = (const float*)d_in[12];
  const float* dt_proj_b = (const float*)d_in[13];
  const float* A_log     = (const float*)d_in[14];
  const float* D_param   = (const float*)d_in[15];
  const float* out_proj_w= (const float*)d_in[16];
  const float* norm_f_w  = (const float*)d_in[17];
  const float* norm_f_b  = (const float*)d_in[18];
  const float* q_w       = (const float*)d_in[19];
  const float* kv_w      = (const float*)d_in[20];
  const float* proj_w    = (const float*)d_in[21];
  const float* proj_b    = (const float*)d_in[22];
  const float* n21_w     = (const float*)d_in[23];
  const float* n21_b     = (const float*)d_in[24];
  const float* n22_w     = (const float*)d_in[25];
  const float* n22_b     = (const float*)d_in[26];
  const float* n2_w      = (const float*)d_in[27];
  const float* n2_b      = (const float*)d_in[28];
  const float* fc1_w     = (const float*)d_in[29];
  const float* fc1_b     = (const float*)d_in[30];
  const float* fc2_w     = (const float*)d_in[31];
  const float* fc2_b     = (const float*)d_in[32];
  const float* head_w    = (const float*)d_in[33];
  const float* head_b    = (const float*)d_in[34];

  float* ws = (float*)d_ws;
  float* hid = ws + O_HID;
  float* res = ws + O_RES;
  float* hn  = ws + O_HN;
  float* xz  = ws + O_XZ;
  float* xs  = ws + O_XS;
  float* dbl = ws + O_DBL;
  float* dtb = ws + O_DT;
  float* yf  = ws + O_YF;
  float* yg  = ws + O_YG;
  // phase aliases
  float* Apat = xz;                   // 3136x768  <= SZ_XZ
  float* Pout = xs;                   // 3136x192  <= SZ_XS
  float* xfin = hn;
  float* qn   = xz;
  float* kvn  = xz + SZ_TOK;
  float* qb   = xz + 2 * SZ_TOK;
  float* kvb  = xs;                   // BL x 2D == SZ_XS
  float* ob   = dtb;
  float* t2   = yf;
  float* g    = xz;
  float* xcls = dbl;

  // ---- patch embedding ----
  {
    const int n = BB * NPp * CH * PP * PP;
    k_im2col<<<(n + 255) / 256, 256, 0, stream>>>(images, Apat);
    gemm(stream, Apat, CH * PP * PP, patch_w, patch_b, nullptr, Pout,
         BB * NPp, DD, CH * PP * PP, 0);
    const int n2 = BB * LL * DD;
    k_assemble<<<(n2 + 255) / 256, 256, 0, stream>>>(Pout, pos_embed, cls_token,
                                                     hid, res);
  }

  // ---- 12 bidirectional Mamba layers ----
  for (int layer = 0; layer < DEPTH; ++layer) {
    // res += hidden ; hn = LN(res)
    k_ln<<<(BL + 7) / 8, 256, 0, stream>>>(res, hid, res,
                                           ln_w + layer * DD, ln_b + layer * DD,
                                           hn, BL);
    // xz = hn @ in_proj_w^T            (3152 x 768, K=192)
    gemm(stream, hn, DD, in_proj_w + (size_t)layer * 2 * EE * DD,
         nullptr, nullptr, xz, BL, 2 * EE, DD, 0);
    // xs = silu(causal_conv(x))        (3152 x 384)
    {
      const int n = BB * LL * EE;
      k_conv_silu<<<(n + 255) / 256, 256, 0, stream>>>(
          xz, conv_w + (size_t)layer * EE * KC, conv_b + (size_t)layer * EE, xs);
    }
    // dbl = xs @ x_proj_w^T            (3152 x 44, K=384)
    gemm(stream, xs, EE, x_proj_w + (size_t)layer * DBLW * EE,
         nullptr, nullptr, dbl, BL, DBLW, EE, 0);
    // dt = softplus(dbl[:, :12] @ dt_proj_w^T + dt_proj_b)  (3152 x 384, K=12)
    gemm(stream, dbl, DBLW, dt_proj_w + (size_t)layer * EE * RR,
         dt_proj_b + (size_t)layer * EE, nullptr, dtb, BL, EE, RR, 2);
    // bidirectional scan + D skip + silu(z) gate
    k_scan<<<BB, EE, 0, stream>>>(dtb, xs, dbl, xz,
                                  A_log + (size_t)layer * EE * NSs,
                                  D_param + (size_t)layer * EE, yf, yg);
    // hidden = yg @ out_proj_w^T       (3152 x 192, K=384)
    gemm(stream, yg, EE, out_proj_w + (size_t)layer * DD * EE,
         nullptr, nullptr, hid, BL, DD, EE, 0);
  }

  // ---- final norm + attention + MLP + head ----
  k_ln<<<(BL + 7) / 8, 256, 0, stream>>>(res, hid, nullptr, norm_f_w, norm_f_b,
                                         xfin, BL);
  k_ln<<<(BL + 7) / 8, 256, 0, stream>>>(xfin, nullptr, nullptr, n21_w, n21_b, qn, BL);
  k_ln<<<(BL + 7) / 8, 256, 0, stream>>>(xfin, nullptr, nullptr, n22_w, n22_b, kvn, BL);
  gemm(stream, qn,  DD, q_w,  nullptr, nullptr, qb,  BL, DD,     DD, 0);
  gemm(stream, kvn, DD, kv_w, nullptr, nullptr, kvb, BL, 2 * DD, DD, 0);
  {
    const int waves = BB * HEADS * LL;
    k_attn<<<(waves + 7) / 8, 256, 0, stream>>>(qb, kvb, mask, ob);
  }
  // x = x + o @ proj_w^T + proj_b  (in place on xfin)
  gemm(stream, ob, DD, proj_w, proj_b, xfin, xfin, BL, DD, DD, 0);
  // MLP
  k_ln<<<(BL + 7) / 8, 256, 0, stream>>>(xfin, nullptr, nullptr, n2_w, n2_b, t2, BL);
  gemm(stream, t2, DD, fc1_w, fc1_b, nullptr, g, BL, FF, DD, 1);      // gelu
  gemm(stream, g, FF, fc2_w, fc2_b, xfin, xfin, BL, DD, FF, 0);       // +resid
  // head on CLS token
  {
    const int n = BB * DD;
    k_gather_cls<<<(n + 255) / 256, 256, 0, stream>>>(xfin, xcls);
  }
  gemm(stream, xcls, DD, head_w, head_b, nullptr, (float*)d_out, BB, NCLS, DD, 0);
}
